// QMultiheadAttention_10866267259187
// MI455X (gfx1250) — 16-bit and 8-bit matrix operand order repaired and checked by reading, compiled, not run on hardware
//
#include <hip/hip_runtime.h>

typedef __attribute__((ext_vector_type(16))) __bf16 v16bf;
typedef __attribute__((ext_vector_type(8)))  float  v8f;
typedef __attribute__((ext_vector_type(8)))  int    v8i;

#define TT 2048
#define BB 2
#define DDIM 1024
#define HH 16
#define DH 64
#define NN 4096
#define NHEAD 32
#define NQB 128

#define OFF_QF   ((size_t)64)
#define SZ_F     ((size_t)NN*DDIM*4)
#define OFF_KF   (OFF_QF+SZ_F)
#define OFF_VF   (OFF_KF+SZ_F)
#define OFF_Q8   (OFF_VF+SZ_F)
#define SZ_8     ((size_t)NN*DDIM)
#define OFF_K8   (OFF_Q8+SZ_8)
#define OFF_V8T  (OFF_K8+SZ_8)
#define OFF_RS   (OFF_V8T+SZ_8)
#define OFF_ATTN OFF_QF

__device__ __forceinline__ v8f wmma_bf16(v16bf a, v16bf b, v8f c) {
  return __builtin_amdgcn_wmma_f32_16x16x32_bf16(false, a, false, b, (short)0, c, false, false);
}
__device__ __forceinline__ v8i wmma_iu8(v8i a, v8i b, v8i c) {
  return __builtin_amdgcn_wmma_i32_16x16x64_iu8(true, a, true, b, c, false, false);
}

__device__ __forceinline__ v16bf load_a_bf16(const float* __restrict__ src, int ld,
                                             int row0, int k0, int lane) {
  int m = lane & 15, h = lane >> 4;
  const float* r = src + (size_t)(row0 + m) * ld + k0 + h * 8;
  const float4* p = (const float4*)r;
  float4 f0 = p[0], f1 = p[1], f2 = p[4], f3 = p[5];
  v16bf a;
  a[0]=(__bf16)f0.x; a[1]=(__bf16)f0.y; a[2]=(__bf16)f0.z; a[3]=(__bf16)f0.w;
  a[4]=(__bf16)f1.x; a[5]=(__bf16)f1.y; a[6]=(__bf16)f1.z; a[7]=(__bf16)f1.w;
  a[8]=(__bf16)f2.x; a[9]=(__bf16)f2.y; a[10]=(__bf16)f2.z; a[11]=(__bf16)f2.w;
  a[12]=(__bf16)f3.x; a[13]=(__bf16)f3.y; a[14]=(__bf16)f3.z; a[15]=(__bf16)f3.w;
  return a;
}
__device__ __forceinline__ v16bf load_a_fq_bf16(const float* __restrict__ src, int ld,
                                                int row0, int k0, int lane,
                                                float inv_s, float s) {
  int m = lane & 15, h = lane >> 4;
  const float* r = src + (size_t)(row0 + m) * ld + k0 + h * 8;
  v16bf a;
#pragma unroll
  for (int i = 0; i < 8; ++i) {
    float v0 = r[i], v1 = r[16 + i];
    int q0 = max(-127, min(127, __float2int_rn(v0 * inv_s)));
    int q1 = max(-127, min(127, __float2int_rn(v1 * inv_s)));
    a[i]     = (__bf16)((float)q0 * s);
    a[8 + i] = (__bf16)((float)q1 * s);
  }
  return a;
}
__device__ __forceinline__ v16bf load_b_bf16(const float* __restrict__ w, int ld,
                                             int col0, int k0, int lane) {
  int n = lane & 15, h = lane >> 4;
  const float4* p = (const float4*)(w + (size_t)(col0 + n) * ld + k0 + h * 8);
  float4 f0 = p[0], f1 = p[1], f2 = p[4], f3 = p[5];
  v16bf b;
  b[0]=(__bf16)f0.x; b[1]=(__bf16)f0.y; b[2]=(__bf16)f0.z; b[3]=(__bf16)f0.w;
  b[4]=(__bf16)f1.x; b[5]=(__bf16)f1.y; b[6]=(__bf16)f1.z; b[7]=(__bf16)f1.w;
  b[8]=(__bf16)f2.x; b[9]=(__bf16)f2.y; b[10]=(__bf16)f2.z; b[11]=(__bf16)f2.w;
  b[12]=(__bf16)f3.x; b[13]=(__bf16)f3.y; b[14]=(__bf16)f3.z; b[15]=(__bf16)f3.w;
  return b;
}
__device__ __forceinline__ v8i load_a_i8(const char* __restrict__ base, int row0, int lane) {
  int m = lane & 15, h = lane >> 4;
  const char* r = base + (size_t)(row0 + m) * 64 + h * 8;
  int2 d0 = *(const int2*)(r);
  int2 d1 = *(const int2*)(r + 16);
  int2 d2 = *(const int2*)(r + 32);
  int2 d3 = *(const int2*)(r + 48);
  v8i a; a[0]=d0.x; a[1]=d0.y; a[2]=d1.x; a[3]=d1.y; a[4]=d2.x; a[5]=d2.y; a[6]=d3.x; a[7]=d3.y;
  return a;
}
__device__ __forceinline__ v8i load_a_lds_i8(const char* lds, int lane) {
  int m = lane & 15, h = lane >> 4;
  const char* r = lds + m * 64 + h * 8;
  int2 d0 = *(const int2*)(r);
  int2 d1 = *(const int2*)(r + 16);
  int2 d2 = *(const int2*)(r + 32);
  int2 d3 = *(const int2*)(r + 48);
  v8i a; a[0]=d0.x; a[1]=d0.y; a[2]=d1.x; a[3]=d1.y; a[4]=d2.x; a[5]=d2.y; a[6]=d3.x; a[7]=d3.y;
  return a;
}
__device__ __forceinline__ v8i load_b_i8(const char* __restrict__ base, size_t rowstride,
                                         int col0, int kbyte0, int lane) {
  int n = lane & 15, h = lane >> 4;
  const char* r = base + (size_t)(col0 + n) * rowstride + kbyte0 + h * 8;
  int2 d0 = *(const int2*)(r);
  int2 d1 = *(const int2*)(r + 16);
  int2 d2 = *(const int2*)(r + 32);
  int2 d3 = *(const int2*)(r + 48);
  v8i b; b[0]=d0.x; b[1]=d0.y; b[2]=d1.x; b[3]=d1.y; b[4]=d2.x; b[5]=d2.y; b[6]=d3.x; b[7]=d3.y;
  return b;
}

__device__ __forceinline__ float wave_max_f(float v) {
#pragma unroll
  for (int m = 16; m >= 1; m >>= 1) v = fmaxf(v, __shfl_xor(v, m, 32));
  return v;
}
__device__ __forceinline__ int wave_max_i(int v) {
#pragma unroll
  for (int m = 16; m >= 1; m >>= 1) v = max(v, __shfl_xor(v, m, 32));
  return v;
}

__global__ void k_init_scalars(unsigned* scal) {
  if (threadIdx.x < 16) scal[threadIdx.x] = 0u;
}

__global__ void __launch_bounds__(128) k_qkv_gemm(
    const float* __restrict__ x,
    const float* __restrict__ wq, const float* __restrict__ bq,
    const float* __restrict__ wk, const float* __restrict__ bk,
    const float* __restrict__ wv, const float* __restrict__ bv,
    char* __restrict__ ws) {
  int lane = threadIdx.x & 31, wid = threadIdx.x >> 5;
  int row0 = blockIdx.y * 32;
  int col0 = blockIdx.x * 256 + wid * 64;
  int which = blockIdx.z;
  const float* W    = (which == 0) ? wq : (which == 1) ? wk : wv;
  const float* bias = (which == 0) ? bq : (which == 1) ? bk : bv;
  float* out = (float*)(ws + ((which == 0) ? OFF_QF : (which == 1) ? OFF_KF : OFF_VF));
  unsigned* scal = (unsigned*)ws;

  v8f acc0[4], acc1[4];
#pragma unroll
  for (int c = 0; c < 4; ++c) {
    acc0[c] = (v8f){0,0,0,0,0,0,0,0};
    acc1[c] = (v8f){0,0,0,0,0,0,0,0};
  }

  for (int k0 = 0; k0 < DDIM; k0 += 32) {
    v16bf a0 = load_a_bf16(x, DDIM, row0,      k0, lane);
    v16bf a1 = load_a_bf16(x, DDIM, row0 + 16, k0, lane);
#pragma unroll
    for (int c = 0; c < 4; ++c) {
      v16bf b = load_b_bf16(W, DDIM, col0 + c * 16, k0, lane);
      acc0[c] = wmma_bf16(a0, b, acc0[c]);
      acc1[c] = wmma_bf16(a1, b, acc1[c]);
    }
  }
  int h = lane >> 4, n = lane & 15;
  float scale = (which == 0) ? 0.125f : 1.0f;
  float lmax = 0.f;
#pragma unroll
  for (int c = 0; c < 4; ++c) {
    float bv_ = bias[col0 + c * 16 + n];
#pragma unroll
    for (int g = 0; g < 8; ++g) {
      float v0 = (acc0[c][g] + bv_) * scale;
      float v1 = (acc1[c][g] + bv_) * scale;
      out[(size_t)(row0 + h * 8 + g) * DDIM + col0 + c * 16 + n] = v0;
      out[(size_t)(row0 + 16 + h * 8 + g) * DDIM + col0 + c * 16 + n] = v1;
      lmax = fmaxf(lmax, fmaxf(fabsf(v0), fabsf(v1)));
    }
  }
  lmax = wave_max_f(lmax);
  if (lane == 0) atomicMax(&scal[which], __float_as_uint(lmax));
}

__global__ void __launch_bounds__(256) k_quantize(char* __restrict__ ws) {
  unsigned* scal = (unsigned*)ws;
  int which = blockIdx.y;
  size_t i = (size_t)blockIdx.x * blockDim.x + threadIdx.x;
  float s = __uint_as_float(scal[which]) / 127.f + 1e-8f;
  float inv_s = 1.0f / s;
  int dh = (int)(i & 63);
  size_t r = i >> 6;
  int t  = (int)(r & (TT - 1));
  int bh = (int)(r >> 11);
  int b = bh >> 4, hh = bh & 15;
  const float* src = (const float*)(ws + ((which == 0) ? OFF_QF : (which == 1) ? OFF_KF : OFF_VF));
  float v = src[((size_t)t * BB + b) * DDIM + hh * 64 + dh];
  int qi = max(-127, min(127, __float2int_rn(v * inv_s)));
  if (which == 2) {
    char* dst = (char*)(ws + OFF_V8T);
    dst[((size_t)bh * 64 + dh) * TT + t] = (char)qi;
  } else {
    char* dst = (char*)(ws + ((which == 0) ? OFF_Q8 : OFF_K8));
    dst[i] = (char)qi;
  }
}

__global__ void __launch_bounds__(128) k_score_absmax(char* __restrict__ ws) {
  int lane = threadIdx.x & 31, wid = threadIdx.x >> 5;
  int gw = blockIdx.x * 4 + wid;
  int head = gw >> 7, qb = gw & 127;
  const char* q8 = (const char*)(ws + OFF_Q8) + (size_t)head * TT * 64;
  const char* k8 = (const char*)(ws + OFF_K8) + (size_t)head * TT * 64;
  v8i a = load_a_i8(q8, qb * 16, lane);
  v8i zero = (v8i){0,0,0,0,0,0,0,0};
  int lmax = 0;
  for (int kt = 0; kt < NQB; ++kt) {
    __builtin_prefetch(k8 + (size_t)(((kt + 8) & 127) * 16) * 64 + lane * 32, 0, 0);
    v8i b = load_b_i8(k8, 64, kt * 16, 0, lane);
    v8i c = wmma_iu8(a, b, zero);
#pragma unroll
    for (int g = 0; g < 8; ++g) { int v = c[g]; lmax = max(lmax, v < 0 ? -v : v); }
  }
  lmax = wave_max_i(lmax);
  if (lane == 0) atomicMax((int*)ws + 3, lmax);
}

__global__ void __launch_bounds__(128) k_rowstats(char* __restrict__ ws) {
  int lane = threadIdx.x & 31, wid = threadIdx.x >> 5;
  int gw = blockIdx.x * 4 + wid;
  int head = gw >> 7, qb = gw & 127;
  unsigned* scal = (unsigned*)ws;
  float sq = __uint_as_float(scal[0]) / 127.f + 1e-8f;
  float sk = __uint_as_float(scal[1]) / 127.f + 1e-8f;
  float sqk = sq * sk;
  float ss = sqk * (float)(((int*)scal)[3]) / 127.f + 1e-8f;
  float c_to_sq = sqk / ss;
  const char* q8 = (const char*)(ws + OFF_Q8) + (size_t)head * TT * 64;
  const char* k8 = (const char*)(ws + OFF_K8) + (size_t)head * TT * 64;
  v8i a = load_a_i8(q8, qb * 16, lane);
  v8i zero = (v8i){0,0,0,0,0,0,0,0};
  float M[8], S[8];
#pragma unroll
  for (int g = 0; g < 8; ++g) { M[g] = -1e30f; S[g] = 0.f; }
  for (int kt = 0; kt < NQB; ++kt) {
    __builtin_prefetch(k8 + (size_t)(((kt + 8) & 127) * 16) * 64 + lane * 32, 0, 0);
    v8i c = wmma_iu8(a, load_b_i8(k8, 64, kt * 16, 0, lane), zero);
#pragma unroll
    for (int g = 0; g < 8; ++g) {
      int si = max(-127, min(127, __float2int_rn((float)c[g] * c_to_sq)));
      float fs = (float)si * ss;
      float nm = fmaxf(M[g], fs);
      S[g] = S[g] * __expf(M[g] - nm) + __expf(fs - nm);
      M[g] = nm;
    }
  }
#pragma unroll
  for (int msk = 8; msk >= 1; msk >>= 1) {
#pragma unroll
    for (int g = 0; g < 8; ++g) {
      float Mo = __shfl_xor(M[g], msk, 32);
      float So = __shfl_xor(S[g], msk, 32);
      float nm = fmaxf(M[g], Mo);
      S[g] = S[g] * __expf(M[g] - nm) + So * __expf(Mo - nm);
      M[g] = nm;
    }
  }
  int h = lane >> 4;
  float2* rs = (float2*)(ws + OFF_RS);
  if ((lane & 15) == 0) {
#pragma unroll
    for (int g = 0; g < 8; ++g)
      rs[(size_t)head * TT + qb * 16 + h * 8 + g] = make_float2(M[g], S[g]);
  }
  float pmax = 0.f;
#pragma unroll
  for (int g = 0; g < 8; ++g) pmax = fmaxf(pmax, 1.0f / S[g]);
  pmax = wave_max_f(pmax);
  if (lane == 0) atomicMax(&scal[4], __float_as_uint(pmax));
}

__global__ void __launch_bounds__(128) k_attn(char* __restrict__ ws) {
  __shared__ __align__(16) char plds[4][1024];
  int lane = threadIdx.x & 31, wid = threadIdx.x >> 5;
  int gw = blockIdx.x * 4 + wid;
  int head = gw >> 7, qb = gw & 127;
  unsigned* scal = (unsigned*)ws;
  float sq = __uint_as_float(scal[0]) / 127.f + 1e-8f;
  float sk = __uint_as_float(scal[1]) / 127.f + 1e-8f;
  float sv = __uint_as_float(scal[2]) / 127.f + 1e-8f;
  float sqk = sq * sk;
  float ss = sqk * (float)(((int*)scal)[3]) / 127.f + 1e-8f;
  float sp = __uint_as_float(scal[4]) / 127.f + 1e-8f;
  float c_to_sq = sqk / ss;
  const char* q8  = (const char*)(ws + OFF_Q8)  + (size_t)head * TT * 64;
  const char* k8  = (const char*)(ws + OFF_K8)  + (size_t)head * TT * 64;
  const char* v8t = (const char*)(ws + OFF_V8T) + (size_t)head * 64 * TT;
  const float2* rs = (const float2*)(ws + OFF_RS);
  int h = lane >> 4, n = lane & 15;

  v8i a = load_a_i8(q8, qb * 16, lane);
  float M[8], invSsp[8];
#pragma unroll
  for (int g = 0; g < 8; ++g) {
    float2 t = rs[(size_t)head * TT + qb * 16 + h * 8 + g];
    M[g] = t.x;
    invSsp[g] = 1.0f / (t.y * sp);
  }
  v8i zero = (v8i){0,0,0,0,0,0,0,0};
  v8i accD[4];
#pragma unroll
  for (int c = 0; c < 4; ++c) accD[c] = zero;
  char* myp = plds[wid];

  for (int kc = 0; kc < TT / 64; ++kc) {
    __builtin_prefetch(k8 + (size_t)(((kc + 2) & 31) * 64) * 64 + lane * 128, 0, 0);
    __builtin_prefetch(v8t + (size_t)(lane & 15) * TT * 4 + (size_t)((kc + 2) & 31) * 64, 0, 0);
#pragma unroll
    for (int kt = 0; kt < 4; ++kt) {
      v8i c = wmma_iu8(a, load_b_i8(k8, 64, kc * 64 + kt * 16, 0, lane), zero);
#pragma unroll
      for (int g = 0; g < 8; ++g) {
        int si = max(-127, min(127, __float2int_rn((float)c[g] * c_to_sq)));
        float fs = (float)si * ss;
        float psp = __expf(fs - M[g]) * invSsp[g];
        int pi = max(0, min(127, __float2int_rn(psp)));
        myp[(h * 8 + g) * 64 + kt * 16 + n] = (char)pi;
      }
    }
    asm volatile("s_wait_dscnt 0" ::: "memory");
    v8i pa = load_a_lds_i8(myp, lane);
#pragma unroll
    for (int c4 = 0; c4 < 4; ++c4) {
      v8i b = load_b_i8(v8t, TT, c4 * 16, kc * 64, lane);
      accD[c4] = wmma_iu8(pa, b, accD[c4]);
    }
  }
  float spv = sp * sv, lmax = 0.f;
  float* attn = (float*)(ws + OFF_ATTN);
  int bb = head >> 4, hh = head & 15;
#pragma unroll
  for (int c4 = 0; c4 < 4; ++c4) {
#pragma unroll
    for (int g = 0; g < 8; ++g) {
      float v = (float)accD[c4][g] * spv;
      int t = qb * 16 + h * 8 + g;
      attn[((size_t)t * BB + bb) * DDIM + hh * 64 + c4 * 16 + n] = v;
      lmax = fmaxf(lmax, fabsf(v));
    }
  }
  lmax = wave_max_f(lmax);
  if (lane == 0) atomicMax(&scal[5], __float_as_uint(lmax));
}

__global__ void __launch_bounds__(128) k_out_gemm(
    const float* __restrict__ wo, const float* __restrict__ bo,
    char* __restrict__ ws, float* __restrict__ out) {
  int lane = threadIdx.x & 31, wid = threadIdx.x >> 5;
  int row0 = blockIdx.y * 32;
  int col0 = blockIdx.x * 256 + wid * 64;
  unsigned* scal = (unsigned*)ws;
  float sa = __uint_as_float(scal[5]) / 127.f + 1e-8f;
  float inv_sa = 1.0f / sa;
  const float* attn = (const float*)(ws + OFF_ATTN);

  v8f acc0[4], acc1[4];
#pragma unroll
  for (int c = 0; c < 4; ++c) {
    acc0[c] = (v8f){0,0,0,0,0,0,0,0};
    acc1[c] = (v8f){0,0,0,0,0,0,0,0};
  }
  for (int k0 = 0; k0 < DDIM; k0 += 32) {
    v16bf a0 = load_a_fq_bf16(attn, DDIM, row0,      k0, lane, inv_sa, sa);
    v16bf a1 = load_a_fq_bf16(attn, DDIM, row0 + 16, k0, lane, inv_sa, sa);
#pragma unroll
    for (int c = 0; c < 4; ++c) {
      v16bf b = load_b_bf16(wo, DDIM, col0 + c * 16, k0, lane);
      acc0[c] = wmma_bf16(a0, b, acc0[c]);
      acc1[c] = wmma_bf16(a1, b, acc1[c]);
    }
  }
  int h = lane >> 4, n = lane & 15;
#pragma unroll
  for (int c = 0; c < 4; ++c) {
    float bv_ = bo[col0 + c * 16 + n];
#pragma unroll
    for (int g = 0; g < 8; ++g) {
      __builtin_nontemporal_store(acc0[c][g] + bv_,
          out + (size_t)(row0 + h * 8 + g) * DDIM + col0 + c * 16 + n);
      __builtin_nontemporal_store(acc1[c][g] + bv_,
          out + (size_t)(row0 + 16 + h * 8 + g) * DDIM + col0 + c * 16 + n);
    }
  }
}

extern "C" void kernel_launch(void* const* d_in, const int* in_sizes, int n_in,
                              void* d_out, int out_size, void* d_ws, size_t ws_size,
                              hipStream_t stream) {
  (void)in_sizes; (void)n_in; (void)out_size; (void)ws_size;
  const float* x  = (const float*)d_in[0];
  const float* wq = (const float*)d_in[1];
  const float* bq = (const float*)d_in[2];
  const float* wk = (const float*)d_in[3];
  const float* bk = (const float*)d_in[4];
  const float* wv = (const float*)d_in[5];
  const float* bv = (const float*)d_in[6];
  const float* wo = (const float*)d_in[7];
  const float* bo = (const float*)d_in[8];
  char* ws = (char*)d_ws;
  float* out = (float*)d_out;

  k_init_scalars<<<1, 32, 0, stream>>>((unsigned*)ws);
  k_qkv_gemm<<<dim3(DDIM / 256, NN / 32, 3), 128, 0, stream>>>(x, wq, bq, wk, bk, wv, bv, ws);
  k_quantize<<<dim3((unsigned)((size_t)NN * DDIM / 256), 3), 256, 0, stream>>>(ws);
  k_score_absmax<<<dim3(NHEAD * NQB / 4), 128, 0, stream>>>(ws);
  k_rowstats<<<dim3(NHEAD * NQB / 4), 128, 0, stream>>>(ws);
  k_attn<<<dim3(NHEAD * NQB / 4), 128, 0, stream>>>(ws);
  k_out_gemm<<<dim3(DDIM / 256, NN / 32), 128, 0, stream>>>(wo, bo, ws, out);
}
